// MptAttention_37254546326160
// MI455X (gfx1250) — matrix operand order repaired and checked by reading, not compiled, not run on hardware
//
#include <hip/hip_runtime.h>
#include <hip/hip_bf16.h>

// ---------------------------------------------------------------------------
// MPT attention for MI455X (gfx1250, wave32, WMMA 16x16x32 f16 -> f32 acc)
// B=2, S=2048, H=2048, NH=16, HD=128
// Fragments laid out so every per-lane load is a contiguous 16B/32B chunk
// -> global_load_b128 / ds_load_b128.  Softmax working set lives entirely in
// CDNA5's 320KB-per-WGP LDS (128KB f32 scores + 64KB f16 P panel), so the
// attn_weights region is written exactly once (pure output, no scratch).
// ---------------------------------------------------------------------------

#define B_   2
#define S_   2048
#define H_   2048
#define NH_  16
#define HD_  128
#define BS_  (B_ * S_)          // 4096
#define H3_  (3 * H_)           // 6144
#define SCALE_ 0.08838834764831845f   // 1/sqrt(128)
#define NEGBIG_ (-3.0e38f)

typedef _Float16 f16;
typedef __attribute__((ext_vector_type(16))) _Float16 v16h;
typedef __attribute__((ext_vector_type(8)))  float    v8f;

union FragU { unsigned u[8]; v16h v; };
union F16x2 { unsigned u; _Float16 h[2]; };

__device__ __forceinline__ v8f wmma16(const FragU& a, const FragU& b, v8f c) {
  // emits v_wmma_f32_16x16x32_f16
  return __builtin_amdgcn_wmma_f32_16x16x32_f16(
      /*neg_a=*/false, a.v, /*neg_b=*/false, b.v,
      /*c_mod=*/(short)0, c, /*reuse_a=*/false, /*reuse_b=*/false);
}

// A-matrix (16x32 f16) per-lane K offset for VGPR v, lane half (0/1).
__device__ __forceinline__ int a_koff(int v, int half) {
  return (v < 4) ? (half * 8 + 2 * v) : (16 + half * 8 + 2 * (v - 4));
}
// B-matrix (32x16 f16) per-lane K offset for VGPR v, lane half.
__device__ __forceinline__ int b_koff(int v, int half) {
  return (v < 4) ? (half * 8 + 2 * v) : (16 + half * 8 + 2 * (v - 4));
}

// ---------------------------------------------------------------------------
// K0a: fp32 -> f16 conversion (grid-stride)
// ---------------------------------------------------------------------------
__global__ __launch_bounds__(256) void cvt_f32_f16(const float* __restrict__ s,
                                                   f16* __restrict__ d, size_t n) {
  size_t i = (size_t)blockIdx.x * blockDim.x + threadIdx.x;
  size_t stride = (size_t)gridDim.x * blockDim.x;
  for (; i < n; i += stride) d[i] = (f16)s[i];
}

// ---------------------------------------------------------------------------
// K0b: fp32 [R,C] -> f16 transposed [C,R], LDS-tiled (32x32), coalesced.
// ---------------------------------------------------------------------------
__global__ __launch_bounds__(256) void cvt_f32_f16_T(const float* __restrict__ src,
                                                     f16* __restrict__ dst,
                                                     int R, int C) {
  __shared__ f16 t[32][33];
  const int r0 = blockIdx.y * 32, c0 = blockIdx.x * 32;
  const int tx = threadIdx.x & 31, ty = threadIdx.x >> 5;   // 32 x 8
#pragma unroll
  for (int i = 0; i < 32; i += 8)
    t[ty + i][tx] = (f16)src[(size_t)(r0 + ty + i) * C + (c0 + tx)];
  __syncthreads();
#pragma unroll
  for (int i = 0; i < 32; i += 8)
    dst[(size_t)(c0 + ty + i) * R + (r0 + tx)] = t[tx][ty + i];
}

// ---------------------------------------------------------------------------
// K0c: build d-major V^T:  vt[b,h,d,s] = qkv[b,s, 2H + h*HD + d]   (f16->f16)
// ---------------------------------------------------------------------------
__global__ __launch_bounds__(256) void transpose_v(const f16* __restrict__ qkv,
                                                   f16* __restrict__ vt) {
  __shared__ f16 t[32][33];
  const int bh = blockIdx.z, b = bh / NH_, h = bh % NH_;
  const int s0 = blockIdx.x * 32, d0 = blockIdx.y * 32;
  const int tx = threadIdx.x & 31, ty = threadIdx.x >> 5;
#pragma unroll
  for (int i = 0; i < 32; i += 8)
    t[ty + i][tx] = qkv[(size_t)(b * S_ + s0 + ty + i) * H3_ + 2 * H_
                        + (size_t)h * HD_ + (d0 + tx)];
  __syncthreads();
#pragma unroll
  for (int i = 0; i < 32; i += 8)
    vt[((size_t)(b * NH_ + h) * HD_ + (d0 + ty + i)) * S_ + (s0 + tx)] = t[tx][ty + i];
}

// ---------------------------------------------------------------------------
// K1: C[M,N] = A[M,K] * Bt[N,K]^T.  A row-major f16, Bt is B TRANSPOSED
// ([N,K] row-major) so the per-lane B fragment (column fixed, K varying) is a
// pair of contiguous 16B chunks -> 2x global_load_b128 per WMMA.
// 256 threads = 8 waves; each wave owns one 16x16 C tile (16M x 128N / block).
// ---------------------------------------------------------------------------
template <bool OUT_F16>
__global__ __launch_bounds__(256) void gemm_f16_nt(const f16* __restrict__ A,
                                                   const f16* __restrict__ Bt,
                                                   void* __restrict__ Cout,
                                                   int M, int N, int K) {
  const int tid  = threadIdx.x;
  const int wv   = tid >> 5;
  const int lane = tid & 31;
  const int lrow = lane & 15;
  const int half = lane >> 4;

  const int nbase = blockIdx.x * 128 + wv * 16;
  const int mbase = blockIdx.y * 16;
  const int ncol  = nbase + lrow;

  v8f acc = {0.f, 0.f, 0.f, 0.f, 0.f, 0.f, 0.f, 0.f};

  const f16* arow = A  + (size_t)(mbase + lrow) * K;
  const f16* brow = Bt + (size_t)ncol * K;

  for (int kk0 = 0; kk0 < K; kk0 += 32) {
    __builtin_prefetch(arow + kk0 + 256, 0, 1);   // global_prefetch_b8
    __builtin_prefetch(brow + kk0 + 256, 0, 1);
    FragU fa, fb;
#pragma unroll
    for (int v = 0; v < 8; ++v) {
      fa.u[v] = *(const unsigned*)(arow + kk0 + a_koff(v, half));
      fb.u[v] = *(const unsigned*)(brow + kk0 + b_koff(v, half));
    }
    acc = wmma16(fa, fb, acc);
  }

#pragma unroll
  for (int r = 0; r < 8; ++r) {
    int m = mbase + r + half * 8;
    size_t idx = (size_t)m * N + ncol;
    if (OUT_F16) ((f16*)Cout)[idx] = (f16)acc[r];
    else         ((float*)Cout)[idx] = acc[r];
  }
}

// ---------------------------------------------------------------------------
// K2: fused attention for one (b,h) and one 16-query tile.
//   phase 1: scores = Q*K^T*scale + bias + causal mask -> LDS f32 panel
//            (fully-masked key tiles skip the WMMA); running row max.
//   phase 2: row sums of exp(s - max) from LDS.
//   phase 3: w = exp(s-max)/sum -> global attn_weights (single write, pure
//            output) and f16 copy -> LDS P panel.
//   phase 4: ctx = P(LDS) * V^T(d-major) via WMMA, K-loop causally truncated.
// LDS: 128KB scores + 64KB P + reductions  (~198KB of the 320KB per WGP).
// ---------------------------------------------------------------------------
__global__ __launch_bounds__(256) void attn_fused(
    const f16* __restrict__ qkv,      // [B,S,3H] f16 (Q|K|V blocks of H cols)
    const f16* __restrict__ vt,       // [B,NH,HD,S] f16 (d-major V)
    const float* __restrict__ bias,   // [NH,S,S] fp32
    float* __restrict__ attw,         // [B,NH,S,S] fp32 (output #2)
    f16* __restrict__ ctx)            // [B,S,H] f16
{
  const int bh = blockIdx.x;                 // 0..B*NH-1
  const int b  = bh / NH_;
  const int h  = bh % NH_;
  const int qbase = blockIdx.y * 16;

  const int tid  = threadIdx.x;
  const int wv   = tid >> 5;
  const int lane = tid & 31;
  const int lrow = lane & 15;
  const int half = lane >> 4;

  __shared__ float sc[16][S_ + 4];    // f32 score panel, +4 row pad (banks)
  __shared__ f16   pw[16][S_ + 8];    // f16 P panel, +16B row pad (banks/align)
  __shared__ float pmax[8][16];
  __shared__ float smax[16];
  __shared__ float ssum[16];

  float* wrow = attw + ((size_t)bh * S_ + qbase) * S_;   // 16 rows x S_

  // ---- Q fragment: A matrix 16 x HD, 4 K-steps of 32 (contiguous in d) ----
  unsigned qf[4][8];
  {
    const f16* qrow = qkv + (size_t)(b * S_ + qbase + lrow) * H3_ + (size_t)h * HD_;
#pragma unroll
    for (int ks = 0; ks < 4; ++ks)
#pragma unroll
      for (int v = 0; v < 8; ++v)
        qf[ks][v] = *(const unsigned*)(qrow + ks * 32 + a_koff(v, half));
  }

  // ---- phase 1: scores -> LDS, running row max ----
  float rmax[8];
#pragma unroll
  for (int r = 0; r < 8; ++r) rmax[r] = NEGBIG_;

  for (int j = wv; j < S_ / 16; j += 8) {
    const int kb = j * 16;
    const int kcol = kb + lrow;
    if (kb > qbase + 15) {
      // fully-masked tile: weights are exactly 0; skip the WMMA
#pragma unroll
      for (int r = 0; r < 8; ++r) sc[r + half * 8][kcol] = NEGBIG_;
      continue;
    }
    // B fragment: B[kk,n] = K[kb+n][kk]; key fixed per lane -> contiguous d.
    const f16* krow = qkv + (size_t)(b * S_ + kb + lrow) * H3_ + (size_t)H_
                      + (size_t)h * HD_;
    v8f acc = {0.f, 0.f, 0.f, 0.f, 0.f, 0.f, 0.f, 0.f};
#pragma unroll
    for (int ks = 0; ks < 4; ++ks) {
      FragU fa, fb;
#pragma unroll
      for (int v = 0; v < 8; ++v) {
        fa.u[v] = qf[ks][v];
        fb.u[v] = *(const unsigned*)(krow + ks * 32 + b_koff(v, half));
      }
      acc = wmma16(fa, fb, acc);
    }
#pragma unroll
    for (int r = 0; r < 8; ++r) {
      int m = r + half * 8;                 // local query row 0..15
      int q = qbase + m;
      float s = acc[r] * SCALE_ + bias[((size_t)h * S_ + q) * S_ + kcol];
      if (kcol > q) s = NEGBIG_;            // causal mask (finfo.min analog)
      sc[m][kcol] = s;
      rmax[r] = fmaxf(rmax[r], s);
    }
  }

  // reduce row max across the 16 lanes of each half (rows stay in-half)
#pragma unroll
  for (int r = 0; r < 8; ++r) {
    float v = rmax[r];
    for (int off = 1; off < 16; off <<= 1) v = fmaxf(v, __shfl_xor(v, off, 32));
    rmax[r] = v;
  }
  if (lrow == 0)
#pragma unroll
    for (int r = 0; r < 8; ++r) pmax[wv][r + half * 8] = rmax[r];
  __syncthreads();

  if (tid < 16) {
    float m = NEGBIG_;
#pragma unroll
    for (int w = 0; w < 8; ++w) m = fmaxf(m, pmax[w][tid]);
    smax[tid] = m;
    ssum[tid] = 0.f;
  }
  __syncthreads();

  // ---- phase 2: row sums from LDS (16 threads per row) ----
  {
    int row = tid >> 4, c0 = tid & 15;
    float m = smax[row], local = 0.f;
    const float* pr = sc[row];
    for (int c = c0; c < S_; c += 16) local += __expf(pr[c] - m);
    atomicAdd(&ssum[row], local);
  }
  __syncthreads();

  // ---- phase 3: normalize; single global write + f16 copy into LDS ----
  {
    int row = tid >> 4, c0 = tid & 15;
    float m = smax[row], inv = 1.0f / ssum[row];
    const float* pr = sc[row];
    float* gw = wrow + (size_t)row * S_;
    for (int c = c0; c < S_; c += 16) {
      float w = __expf(pr[c] - m) * inv;
      gw[c] = w;                   // attn_weights output (only global write)
      pw[row][c] = (f16)w;         // P panel for the WMMA below
    }
  }
  __syncthreads();

  // ---- phase 4: ctx tile = P(LDS) * V(S x HD); wave wv -> 16-wide d slice --
  {
    const int dbase = wv * 16;
    v8f acc = {0.f, 0.f, 0.f, 0.f, 0.f, 0.f, 0.f, 0.f};
    const f16* prow = pw[lrow];                                   // A row
    const f16* vrow = vt + ((size_t)(b * NH_ + h) * HD_ + (dbase + lrow)) * S_;
    const int ks_end = (qbase + 16 + 31) / 32;   // causal: P is 0 beyond this
    for (int ks = 0; ks < ks_end; ++ks) {
      FragU fa, fb;
#pragma unroll
      for (int v = 0; v < 8; ++v) {
        fa.u[v] = *(const unsigned*)(prow + ks * 32 + a_koff(v, half)); // ds_load
        fb.u[v] = *(const unsigned*)(vrow + ks * 32 + b_koff(v, half)); // b128
      }
      acc = wmma16(fa, fb, acc);
    }
#pragma unroll
    for (int r = 0; r < 8; ++r) {
      int q = qbase + r + half * 8;
      ctx[(size_t)(b * S_ + q) * H_ + (size_t)h * HD_ + (dbase + lrow)] = (f16)acc[r];
    }
  }
}

// ---------------------------------------------------------------------------
// Host-side launcher
// ---------------------------------------------------------------------------
extern "C" void kernel_launch(void* const* d_in, const int* in_sizes, int n_in,
                              void* d_out, int out_size, void* d_ws, size_t ws_size,
                              hipStream_t stream) {
  const float* hs   = (const float*)d_in[0];   // [B,S,H]
  const float* pb   = (const float*)d_in[1];   // [NH,S,S]
  // d_in[2] = attention_mask (causal) — reproduced analytically in-kernel
  const float* wqkv = (const float*)d_in[3];   // [H,3H]
  const float* wout = (const float*)d_in[4];   // [H,H]

  float* out_attn = (float*)d_out;                               // [B,S,H]
  float* out_w    = (float*)d_out + (size_t)BS_ * H_;            // [B,NH,S,S]

  // workspace carve-up (all f16)
  char* ws = (char*)d_ws;
  f16* Xb     = (f16*)ws;                ws += (size_t)BS_ * H_  * sizeof(f16);
  f16* WqkvT  = (f16*)ws;                ws += (size_t)H3_ * H_  * sizeof(f16);  // [3H, H]
  f16* WoutT  = (f16*)ws;                ws += (size_t)H_  * H_  * sizeof(f16);  // [H, H]^T
  f16* qkvb   = (f16*)ws;                ws += (size_t)BS_ * H3_ * sizeof(f16);
  f16* Vt     = (f16*)ws;                ws += (size_t)BS_ * H_  * sizeof(f16);  // [B,NH,HD,S]
  f16* ctxb   = (f16*)ws;                ws += (size_t)BS_ * H_  * sizeof(f16);

  // K0: activations to f16; weights to TRANSPOSED f16 ([N,K] row-major)
  cvt_f32_f16<<<2048, 256, 0, stream>>>(hs, Xb, (size_t)BS_ * H_);
  cvt_f32_f16_T<<<dim3(H3_ / 32, H_ / 32), 256, 0, stream>>>(wqkv, WqkvT, H_, H3_);
  cvt_f32_f16_T<<<dim3(H_ / 32, H_ / 32), 256, 0, stream>>>(wout, WoutT, H_, H_);

  // K1: QKV projection  [4096,2048] x [2048,6144] -> f16 qkv
  gemm_f16_nt<true><<<dim3(H3_ / 128, BS_ / 16), 256, 0, stream>>>(
      Xb, WqkvT, (void*)qkvb, BS_, H3_, H_);

  // K0c: d-major V^T for the P*V stage
  transpose_v<<<dim3(S_ / 32, HD_ / 32, B_ * NH_), 256, 0, stream>>>(qkvb, Vt);

  // K2: fused scores + softmax + P*V  (writes attn_weights output once)
  attn_fused<<<dim3(B_ * NH_, S_ / 16), 256, 0, stream>>>(qkvb, Vt, pb, out_w, ctxb);

  // K3: output projection  [4096,2048] x [2048,2048] -> fp32 attn_output
  gemm_f16_nt<false><<<dim3(H_ / 128, BS_ / 16), 256, 0, stream>>>(
      ctxb, WoutT, (void*)out_attn, BS_, H_, H_);
}
